// KLParches_72018011619817
// MI455X (gfx1250) — compile-verified, matrix operand order checked by reading, not run on hardware
//
#include <hip/hip_runtime.h>
#include <hip/hip_bf16.h>

typedef __attribute__((ext_vector_type(16))) _Float16 v16h;
typedef __attribute__((ext_vector_type(8)))  float    v8f;

// ---------------------------------------------------------------------------
// Workspace layout (int32):
//   ws[0   .. 255] : hist1[32][8]   per-sample argmax histograms
//   ws[256 .. 511] : histT[2][16][8] per-(buffer, x-value) tri histograms
// ---------------------------------------------------------------------------

__global__ void kl_zero_ws(int* __restrict__ ws) {
    const int t = threadIdx.x;          // blockDim.x == 256
    ws[t]       = 0;
    ws[t + 256] = 0;
}

// 64-bit wave32 shuffle helpers (two 32-bit lane moves)
__device__ __forceinline__ unsigned long long shfl_down_u64(unsigned long long v, int off) {
    unsigned lo = (unsigned)v, hi = (unsigned)(v >> 32);
    lo = __shfl_down(lo, off, 32);
    hi = __shfl_down(hi, off, 32);
    return ((unsigned long long)hi << 32) | lo;
}
__device__ __forceinline__ unsigned long long shfl_xor_u64(unsigned long long v, int mask) {
    unsigned lo = (unsigned)v, hi = (unsigned)(v >> 32);
    lo = __shfl_xor(lo, mask, 32);
    hi = __shfl_xor(hi, mask, 32);
    return ((unsigned long long)hi << 32) | lo;
}

// ---------------------------------------------------------------------------
// Stage 1: argmax over C=8 + per-sample 8-bin histogram.
// inputs layout [32][8][65536]; grid (64, 32), 256 threads; each thread owns
// 4 consecutive positions -> one float4 load per class (8 x B128, coalesced).
// Counts packed 8 bins x 8 bits in a u64; wave-reduced; 8 atomics per wave.
// ---------------------------------------------------------------------------
__global__ void kl_argmax_hist(const float* __restrict__ in, int* __restrict__ hist1) {
    const int sample = blockIdx.y;
    const int tid    = blockIdx.x * blockDim.x + threadIdx.x;  // 0..16383
    const int p0     = tid * 4;
    const float* base = in + (size_t)sample * (8u * 65536u);

    float4 v[8];
#pragma unroll
    for (int c = 0; c < 8; ++c)
        v[c] = *(const float4*)(base + c * 65536 + p0);

    unsigned long long acc = 0ull;
#pragma unroll
    for (int j = 0; j < 4; ++j) {
        float vv[8];
#pragma unroll
        for (int c = 0; c < 8; ++c) vv[c] = (&v[c].x)[j];
        float m = vv[0];
#pragma unroll
        for (int c = 1; c < 8; ++c) m = fmaxf(m, vv[c]);
        int idx = 7;                      // first index achieving the max
#pragma unroll
        for (int c = 6; c >= 0; --c) idx = (vv[c] == m) ? c : idx;
        acc += 1ull << (idx * 8);         // field max 4/lane, 128/wave: no carry
    }

#pragma unroll
    for (int off = 16; off > 0; off >>= 1)
        acc += shfl_down_u64(acc, off);

    if ((threadIdx.x & 31) == 0) {
#pragma unroll
        for (int b = 0; b < 8; ++b)
            atomicAdd(&hist1[sample * 8 + b], (int)((acc >> (8 * b)) & 0xFFull));
    }
}

// ---------------------------------------------------------------------------
// Stage 2: histograms of tri2/tri3 keyed by (buffer, last-dim index, bin).
// Reads each buffer exactly once, fully coalesced (instead of 32 strided
// per-sample gathers with 16x cache-line amplification).
// grid (512, 2), 256 threads; stride 131072 is a multiple of 16, so each
// lane's x-value is fixed: xv = lane % 16. 64 elements/thread -> packed u64.
// ---------------------------------------------------------------------------
__global__ void kl_tri_hist(const int* __restrict__ tri2, const int* __restrict__ tri3,
                            int* __restrict__ histT) {
    __shared__ int lhist[128];            // [16 xv][8 bins]
    const int buf = blockIdx.y;
    const int tid = threadIdx.x;
    if (tid < 128) lhist[tid] = 0;
    __syncthreads();

    const int t = blockIdx.x * 256 + tid; // 0..131071
    const int* __restrict__ base = (buf == 0) ? tri2 : tri3;
    const int STRIDE = 512 * 256;         // 131072, multiple of 16

    unsigned long long acc = 0ull;
#pragma unroll 4
    for (int it = 0; it < 64; ++it) {
        if (it + 8 < 64)
            __builtin_prefetch(base + t + (it + 8) * STRIDE, 0, 0); // global_prefetch_b8
        const int bin = base[t + it * STRIDE] & 7;
        acc += 1ull << (bin * 8);         // <= 64 per field
    }

    // combine the two lanes sharing each xv (lane L and L+16): <=128 per field
    acc += shfl_xor_u64(acc, 16);

    const int lane = tid & 31;
    if (lane < 16) {
        const int xv = lane;              // == t % 16 for these lanes
#pragma unroll
        for (int b = 0; b < 8; ++b)
            atomicAdd(&lhist[xv * 8 + b], (int)((acc >> (8 * b)) & 0xFFull)); // ds_add
    }
    __syncthreads();
    if (tid < 128) atomicAdd(&histT[buf * 128 + tid], lhist[tid]);
}

// ---------------------------------------------------------------------------
// Stage 3: KL. One wave32, EXEC all-ones. Lane i = sample i computes its 8
// terms exp(disT)*(disT-dis1); the 256 terms are summed with a single
// v_wmma_f32_16x16x32_f16: terms fill the A tile bijectively, B = all-ones,
// so D[m][n] = rowsum_m and total = sum_m rowsum_m = column-0 sums of the
// two lane halves.
// ---------------------------------------------------------------------------
__global__ void kl_final(const int* __restrict__ ws, const int* __restrict__ edad,
                         const int* __restrict__ x, float* __restrict__ out) {
    const int i = threadIdx.x;            // 0..31 (sample)
    const int* hist1 = ws;
    const int* histT = ws + 256;
    const int xv  = x[i] & 15;
    const int buf = (edad[i] <= 24) ? 0 : 1;

    v16h a;
#pragma unroll
    for (int b = 0; b < 8; ++b) {
        const int c1 = hist1[i * 8 + b];
        const int c2 = histT[buf * 128 + xv * 8 + b];
        const float d1 = logf(((c1 > 0) ? (float)c1 : 1e-5f) * (1.0f / 65536.0f));
        const float dT = logf(((c2 > 0) ? (float)c2 : 1e-5f) * (1.0f / 524288.0f));
        const float term = expf(dT) * (dT - d1);
        a[b]     = (_Float16)term;
        a[b + 8] = (_Float16)0.0f;
    }
    v16h ones;
#pragma unroll
    for (int k = 0; k < 16; ++k) ones[k] = (_Float16)1.0f;

    v8f c = {};
    c = __builtin_amdgcn_wmma_f32_16x16x32_f16(
        /*neg_a=*/false, a, /*neg_b=*/false, ones,
        /*c_mod=*/(short)0, c, /*reuse_a=*/false, /*reuse_b=*/false);

    float s = 0.0f;
#pragma unroll
    for (int k = 0; k < 8; ++k) s += c[k];  // column n = lane%16 == 0 on lanes 0 & 16
    s += __shfl_xor(s, 16, 32);             // rows 0-7 + rows 8-15

    if (i == 0) out[0] = s * (2.5f / 256.0f);
}

// ---------------------------------------------------------------------------
extern "C" void kernel_launch(void* const* d_in, const int* in_sizes, int n_in,
                              void* d_out, int out_size, void* d_ws, size_t ws_size,
                              hipStream_t stream) {
    const float* inputs = (const float*)d_in[0];
    const int*   edad   = (const int*)d_in[1];
    // d_in[2] = targets: unused by the reference computation
    const int*   x      = (const int*)d_in[3];
    const int*   tri2   = (const int*)d_in[4];
    const int*   tri3   = (const int*)d_in[5];
    float*       out    = (float*)d_out;
    int*         ws     = (int*)d_ws;

    hipLaunchKernelGGL(kl_zero_ws,     dim3(1),       dim3(256), 0, stream, ws);
    hipLaunchKernelGGL(kl_argmax_hist, dim3(64, 32),  dim3(256), 0, stream, inputs, ws);
    hipLaunchKernelGGL(kl_tri_hist,    dim3(512, 2),  dim3(256), 0, stream, tri2, tri3, ws + 256);
    hipLaunchKernelGGL(kl_final,       dim3(1),       dim3(32),  0, stream, ws, edad, x, out);
}
